// EntityClassifyHeteroAPI_1331439862169
// MI455X (gfx1250) — matrix operand order repaired and checked by reading, not compiled, not run on hardware
//
#include <hip/hip_runtime.h>
#include <hip/hip_bf16.h>

// ---------------------------------------------------------------------------
// RGCN (EntityClassify hetero) on MI455X / gfx1250.
//   layer0: h0 = relu(scatter_sum(embed[src], dst) + b0)            (no weight)
//   layer1: h1 = relu(Σ_r (scatter_sum_r(h0[src_r], dst_r)) @ W1_r + b1)
//   layer2: out =      Σ_r (scatter_sum_r(h1[src_r], dst_r)) @ W2_r + b2
// Dense GEMMs: v_wmma_f32_16x16x32_bf16, bf16 inputs pre-packed; K-loop is
// software-pipelined (double-buffered fragments) so loads overlap WMMAs.
// ---------------------------------------------------------------------------

#define RG_N   50000
#define RG_H   256
#define RG_O   64
#define RG_R   8
#define RG_E   65536

typedef __attribute__((ext_vector_type(16))) __bf16 v16bf;
typedef __attribute__((ext_vector_type(8)))  float  v8f;

union FragBF { v16bf v; unsigned int d[8]; uint4 q[2]; };

__device__ __forceinline__ unsigned int rg_f32_to_bf16(float x) {
    unsigned int u = __float_as_uint(x);
    u = (u + 0x7FFFu + ((u >> 16) & 1u)) >> 16;   // round-to-nearest-even
    return u & 0xFFFFu;
}

__device__ __forceinline__ unsigned int rg_pack2(float lo, float hi) {
    return rg_f32_to_bf16(lo) | (rg_f32_to_bf16(hi) << 16);
}

// A-fragment: 16 contiguous K values -> q0 at +0, q1 at +8 ushorts (+koff).
__device__ __forceinline__ void rg_load_a(FragBF& f, const unsigned short* p) {
    f.q[0] = *(const uint4*)(p);
    f.q[1] = *(const uint4*)(p + 8);
}

// B-fragment: 16 contiguous K values -> q0 at +0, q1 at +8 ushorts.
__device__ __forceinline__ void rg_load_b(FragBF& f, const unsigned short* p) {
    f.q[0] = *(const uint4*)(p);
    f.q[1] = *(const uint4*)(p + 8);
}

// ---------------------------------------------------------------------------
// Zero a float buffer (float4 granularity; all sizes here are /4).
// ---------------------------------------------------------------------------
__global__ void rg_zero_kernel(float4* __restrict__ p, int n4) {
    int i = blockIdx.x * blockDim.x + threadIdx.x;
    if (i < n4) p[i] = make_float4(0.f, 0.f, 0.f, 0.f);
}

// ---------------------------------------------------------------------------
// Edge scatter:  Out[dst[e], :] += H[src[e], :]   (one wave32 per edge,
// 8 floats per lane -> global_atomic_add_f32, no return).
// ---------------------------------------------------------------------------
__global__ void rg_scatter_kernel(const float* __restrict__ H,
                                  const int* __restrict__ src,
                                  const int* __restrict__ dst,
                                  float* __restrict__ Out,
                                  int nEdges) {
    int wave = (int)((blockIdx.x * blockDim.x + threadIdx.x) >> 5);
    int lane = threadIdx.x & 31;
    if (wave >= nEdges) return;
    int s = src[wave];
    int d = dst[wave];
    const float4* hp = (const float4*)(H + (size_t)s * RG_H);
    float*        op = Out + (size_t)d * RG_H;
    float4 v0 = hp[lane * 2 + 0];
    float4 v1 = hp[lane * 2 + 1];
    int c = lane * 8;
    atomicAdd(op + c + 0, v0.x);
    atomicAdd(op + c + 1, v0.y);
    atomicAdd(op + c + 2, v0.z);
    atomicAdd(op + c + 3, v0.w);
    atomicAdd(op + c + 4, v1.x);
    atomicAdd(op + c + 5, v1.y);
    atomicAdd(op + c + 6, v1.z);
    atomicAdd(op + c + 7, v1.w);
}

// ---------------------------------------------------------------------------
// Bias (+ optional ReLU), elementwise over [N, F], F power of two.
// ---------------------------------------------------------------------------
__global__ void rg_bias_act_kernel(float* __restrict__ Hm,
                                   const float* __restrict__ b,
                                   int total, int fmask, int relu) {
    int i = blockIdx.x * blockDim.x + threadIdx.x;
    if (i >= total) return;
    float v = Hm[i] + b[i & fmask];
    Hm[i] = relu ? fmaxf(v, 0.f) : v;
}

// ---------------------------------------------------------------------------
// Pack f32 -> bf16 (RNE), 4 elements per thread (one memory-bound pass).
// ---------------------------------------------------------------------------
__global__ void rg_pack_bf16_kernel(const float4* __restrict__ in,
                                    uint2* __restrict__ out, int n4) {
    int i = blockIdx.x * blockDim.x + threadIdx.x;
    if (i >= n4) return;
    float4 f = in[i];
    out[i] = make_uint2(rg_pack2(f.x, f.y), rg_pack2(f.z, f.w));
}

// ---------------------------------------------------------------------------
// Convert + transpose weights:  W[r, k, n] f32  ->  Wt[r, n, k] bf16.
// ---------------------------------------------------------------------------
__global__ void rg_wt_kernel(const float* __restrict__ W,
                             unsigned short* __restrict__ Wt,
                             int K, int HOUT, int total) {
    int i = blockIdx.x * blockDim.x + threadIdx.x;
    if (i >= total) return;
    int n = i % HOUT;
    int t = i / HOUT;
    int k = t % K;
    int r = t / K;
    Wt[((size_t)r * HOUT + n) * K + k] = (unsigned short)rg_f32_to_bf16(W[i]);
}

// ---------------------------------------------------------------------------
// WMMA GEMM:  C[N, HOUT] += Abf[N, 256] @ Wt^T   (both bf16, Wt is [HOUT,256]).
// One wave32 computes a 16x64 output strip: one A fragment per 32-K step feeds
// 4 v_wmma_f32_16x16x32_bf16. Fragments are double-buffered so the loads for
// step s+1 issue before the WMMAs of step s (loads overlap matrix ops).
//
// A-fragment (16-bit A, 16x32): lane m = lane&15; K-halves split at lane 16
// (koff = (lane>=16)*16); contiguous-K pairs -> two uint4 loads.
// B-fragment: lane n = lane&15, k = (lane>=16)*16 + 0..15 contiguous in the
// pre-transposed Wt -> two uint4 loads.
// ---------------------------------------------------------------------------
template<int HOUT>
__global__ void rg_gemm_bf16_kernel(const unsigned short* __restrict__ Abf,
                                    const unsigned short* __restrict__ Wt,
                                    float* __restrict__ C,
                                    int N) {
    constexpr int TN_TOTAL = HOUT / 16;           // 16 or 4 column tiles
    constexpr int WAVES_PER_ROW = TN_TOTAL / 4;   // 4 tiles per wave
    const int wave = (int)((blockIdx.x * blockDim.x + threadIdx.x) >> 5);
    const int lane = threadIdx.x & 31;
    const int tm  = wave / WAVES_PER_ROW;
    const int tn0 = (wave % WAVES_PER_ROW) * 4;   // first of 4 column tiles
    if (tm * 16 >= N) return;                     // wave-uniform; EXEC stays full

    const int half = lane >> 4;                   // 0 | 1
    const int l15  = lane & 15;
    const int m    = tm * 16 + l15;               // A row for this lane

    const unsigned short* arow = Abf + (size_t)m * RG_H + half * 16;
    const unsigned short* br0  = Wt + (size_t)((tn0 + 0) * 16 + l15) * RG_H + half * 16;
    const unsigned short* br1  = Wt + (size_t)((tn0 + 1) * 16 + l15) * RG_H + half * 16;
    const unsigned short* br2  = Wt + (size_t)((tn0 + 2) * 16 + l15) * RG_H + half * 16;
    const unsigned short* br3  = Wt + (size_t)((tn0 + 3) * 16 + l15) * RG_H + half * 16;

    v8f acc[4] = {};
    FragBF aBuf[2], bBuf[2][4];

    // prologue: stage K-step 0
    rg_load_a(aBuf[0], arow);
    rg_load_b(bBuf[0][0], br0);
    rg_load_b(bBuf[0][1], br1);
    rg_load_b(bBuf[0][2], br2);
    rg_load_b(bBuf[0][3], br3);

    #pragma unroll
    for (int s = 0; s < RG_H / 32; ++s) {
        const int cur = s & 1;
        const int nxt = cur ^ 1;
        if (s < RG_H / 32 - 1) {                  // prefetch K-step s+1
            const int kk = (s + 1) * 32;
            rg_load_a(aBuf[nxt], arow + kk);
            rg_load_b(bBuf[nxt][0], br0 + kk);
            rg_load_b(bBuf[nxt][1], br1 + kk);
            rg_load_b(bBuf[nxt][2], br2 + kk);
            rg_load_b(bBuf[nxt][3], br3 + kk);
        }
        #pragma unroll
        for (int j = 0; j < 4; ++j) {
            acc[j] = __builtin_amdgcn_wmma_f32_16x16x32_bf16(
                         /*neg_a=*/false, aBuf[cur].v,
                         /*neg_b=*/false, bBuf[cur][j].v,
                         /*c_mod=*/(short)0, acc[j],
                         /*reuse_a=*/false, /*reuse_b=*/false);
        }
    }

    // C/D layout: col = lane&15, row = vgpr + 8*(lane>=16).
    #pragma unroll
    for (int j = 0; j < 4; ++j) {
        #pragma unroll
        for (int v = 0; v < 8; ++v) {
            int row = tm * 16 + v + half * 8;
            C[(size_t)row * HOUT + (tn0 + j) * 16 + l15] += acc[j][v];
        }
    }
}

// ---------------------------------------------------------------------------
// Host-side orchestration (all on `stream`, graph-capture safe).
// ---------------------------------------------------------------------------
extern "C" void kernel_launch(void* const* d_in, const int* in_sizes, int n_in,
                              void* d_out, int out_size, void* d_ws, size_t ws_size,
                              hipStream_t stream) {
    const float* embed = (const float*)d_in[0];   // [N, 256]
    const int*   src   = (const int*)  d_in[1];   // [R, E]
    const int*   dst   = (const int*)  d_in[2];   // [R, E]
    const float* W1    = (const float*)d_in[3];   // [R, 256, 256]
    const float* b0    = (const float*)d_in[4];   // [256]
    const float* b1    = (const float*)d_in[5];   // [256]
    const float* W2    = (const float*)d_in[6];   // [R, 256, 64]
    const float* b2    = (const float*)d_in[7];   // [64]
    float*       out   = (float*)d_out;           // [N, 64]

    const size_t NH_BYTES = (size_t)RG_N * RG_H * sizeof(float);     // 51.2 MB
    const size_t NH_BF    = (size_t)RG_N * RG_H * sizeof(unsigned short);
    char* ws = (char*)d_ws;
    float*          h0  = (float*)(ws);
    float*          h1  = (float*)(ws + NH_BYTES);
    float*          Af  = (float*)(ws + 2 * NH_BYTES);               // f32 accum
    unsigned short* Abf = (unsigned short*)(ws + 3 * NH_BYTES);      // bf16 pack
    unsigned short* W1t = (unsigned short*)(ws + 3 * NH_BYTES + NH_BF);
    unsigned short* W2t = (unsigned short*)(ws + 3 * NH_BYTES + NH_BF
                                               + (size_t)RG_R * RG_H * RG_H * 2);

    const int NH = RG_N * RG_H;                   // 12.8M floats
    const int NO = RG_N * RG_O;                   // 3.2M floats
    const int B  = 256;

    // ---- layer 0: h0 = relu(scatter(embed) + b0) --------------------------
    rg_zero_kernel<<<(NH / 4 + B - 1) / B, B, 0, stream>>>((float4*)h0, NH / 4);
    {
        int edges = RG_R * RG_E;                  // all relations at once
        rg_scatter_kernel<<<(edges * 32 + B - 1) / B, B, 0, stream>>>(
            embed, src, dst, h0, edges);
    }
    rg_bias_act_kernel<<<(NH + B - 1) / B, B, 0, stream>>>(h0, b0, NH, RG_H - 1, 1);

    // ---- weight convert/transpose to bf16 ---------------------------------
    {
        int t1 = RG_R * RG_H * RG_H;
        rg_wt_kernel<<<(t1 + B - 1) / B, B, 0, stream>>>(W1, W1t, RG_H, RG_H, t1);
        int t2 = RG_R * RG_H * RG_O;
        rg_wt_kernel<<<(t2 + B - 1) / B, B, 0, stream>>>(W2, W2t, RG_H, RG_O, t2);
    }

    // ---- layer 1: h1 = relu(Σ_r A_r @ W1_r + b1) --------------------------
    rg_zero_kernel<<<(NH / 4 + B - 1) / B, B, 0, stream>>>((float4*)h1, NH / 4);
    for (int r = 0; r < RG_R; ++r) {
        rg_zero_kernel<<<(NH / 4 + B - 1) / B, B, 0, stream>>>((float4*)Af, NH / 4);
        rg_scatter_kernel<<<(RG_E * 32 + B - 1) / B, B, 0, stream>>>(
            h0, src + (size_t)r * RG_E, dst + (size_t)r * RG_E, Af, RG_E);
        rg_pack_bf16_kernel<<<(NH / 4 + B - 1) / B, B, 0, stream>>>(
            (const float4*)Af, (uint2*)Abf, NH / 4);
        int waves = (RG_N / 16) * (RG_H / 64);    // 12500 (16x64 strip / wave)
        rg_gemm_bf16_kernel<RG_H><<<(waves * 32 + B - 1) / B, B, 0, stream>>>(
            Abf, W1t + (size_t)r * RG_H * RG_H, h1, RG_N);
    }
    rg_bias_act_kernel<<<(NH + B - 1) / B, B, 0, stream>>>(h1, b1, NH, RG_H - 1, 1);

    // ---- layer 2: out = Σ_r A_r @ W2_r + b2 -------------------------------
    rg_zero_kernel<<<(NO / 4 + B - 1) / B, B, 0, stream>>>((float4*)out, NO / 4);
    for (int r = 0; r < RG_R; ++r) {
        rg_zero_kernel<<<(NH / 4 + B - 1) / B, B, 0, stream>>>((float4*)Af, NH / 4);
        rg_scatter_kernel<<<(RG_E * 32 + B - 1) / B, B, 0, stream>>>(
            h1, src + (size_t)r * RG_E, dst + (size_t)r * RG_E, Af, RG_E);
        rg_pack_bf16_kernel<<<(NH / 4 + B - 1) / B, B, 0, stream>>>(
            (const float4*)Af, (uint2*)Abf, NH / 4);
        int waves = (RG_N / 16) * (RG_O / 64);    // 3125
        rg_gemm_bf16_kernel<RG_O><<<(waves * 32 + B - 1) / B, B, 0, stream>>>(
            Abf, W2t + (size_t)r * RG_O * RG_H, out, RG_N);
    }
    rg_bias_act_kernel<<<(NO + B - 1) / B, B, 0, stream>>>(out, b2, NO, RG_O - 1, 0);
}
